// GNNLayer_57638461112891
// MI455X (gfx1250) — matrix operand order repaired and checked by reading, not compiled, not run on hardware
//
#include <hip/hip_runtime.h>
#include <hip/hip_bf16.h>

// ---------------- problem constants (from reference) ----------------
#define NN   100000   // nodes
#define NE   1600000  // edges
#define INC  128
#define HIDC 64
#define OUTC 64
#define LSTMH 96
#define NGATE 384     // 4*LSTM_H
#define NG   64       // graphs
#define BN_EPS 1e-5f

// ---------------- WMMA types ----------------
typedef __attribute__((ext_vector_type(16))) __bf16 v16bf_t;
typedef __attribute__((ext_vector_type(8)))  float  v8f_t;

union FragBF { v16bf_t v; unsigned int u[8]; };
union FragF  { v8f_t   v; float f[8]; };

__device__ __forceinline__ unsigned short f2bf(float f) {
  unsigned int u = __float_as_uint(f);
  unsigned int r = 0x7FFFu + ((u >> 16) & 1u);   // round-to-nearest-even
  return (unsigned short)((u + r) >> 16);
}
__device__ __forceinline__ unsigned int pack2bf(float lo, float hi) {
  return ((unsigned int)f2bf(hi) << 16) | (unsigned int)f2bf(lo);
}
__device__ __forceinline__ float sigm(float x) { return 1.0f / (1.0f + __expf(-x)); }

// ---------------- generic bf16 WMMA GEMM ----------------
// C[N x M] = A[N x K] @ B (+ Cin) (+ bias1 + bias2)
// B is (K x M) row-major if transB==0, or (M x K) row-major if transB==1.
// Requires: K % 32 == 0, M % 64 == 0.
// Block tile: 128 rows x 64 cols; 8 wave32 waves in 4x2; each wave: 2x2 WMMA tiles (32x32 of C).
#define TILE_N 128
#define TILE_M 64
#define TILE_K 32
#define KP     36   // LDS row pitch in ushorts (72B: 8B-aligned rows, 18-bank stride)

__global__ __launch_bounds__(256) void gemm_bf16_wmma(
    const float* __restrict__ A, const float* __restrict__ B,
    const float* __restrict__ bias1, const float* __restrict__ bias2,
    const float* __restrict__ Cin, float* __restrict__ C,
    int N, int K, int M, int transB)
{
  __shared__ unsigned short As[TILE_N][KP];   // [row][k]  bf16 bits (9216 B)
  __shared__ unsigned short Bs[TILE_M][KP];   // [m][k]    staged transposed (4608 B)

  const int tid  = threadIdx.x;
  const int wave = tid >> 5;
  const int lane = tid & 31;
  const int l15  = lane & 15;
  const int g    = lane >> 4;

  const int rowBlock = blockIdx.x * TILE_N;
  const int colBlock = blockIdx.y * TILE_M;
  const int wRow = (wave >> 1) * 32;   // 0,32,64,96
  const int wCol = (wave & 1) * 32;    // 0,32

  FragF acc[2][2];
#pragma unroll
  for (int i = 0; i < 2; ++i)
#pragma unroll
    for (int j = 0; j < 2; ++j)
#pragma unroll
      for (int r = 0; r < 8; ++r) acc[i][j].f[r] = 0.0f;

  for (int k0 = 0; k0 < K; k0 += TILE_K) {
    // ---- stage A tile (128 rows x 32 k), float4 loads, branchless row clamp ----
#pragma unroll
    for (int it = 0; it < 4; ++it) {
      int j  = it * 256 + tid;        // 0..1023 quads
      int r  = j >> 3;                // row 0..127
      int kq = (j & 7) * 4;           // 0,4,...,28
      int gr = rowBlock + r;
      int cr = gr < N ? gr : (N - 1); // clamp -> load always in-range
      const float* ap = A + (long long)cr * K + (k0 + kq);
      float4 av = *(const float4*)ap;
      float msk = (gr < N) ? 1.0f : 0.0f;
      *(unsigned int*)&As[r][kq]     = pack2bf(av.x * msk, av.y * msk);
      *(unsigned int*)&As[r][kq + 2] = pack2bf(av.z * msk, av.w * msk);
      __builtin_prefetch(ap + TILE_K, 0, 1);   // speculative; next K-chunk
    }
    // ---- stage B tile (32 k x 64 m) into Bs[m][k] ----
    if (transB) {   // B is M x K: contiguous along k
#pragma unroll
      for (int it = 0; it < 2; ++it) {
        int j  = it * 256 + tid;      // 0..511 quads
        int m  = j >> 3;              // 0..63
        int kq = (j & 7) * 4;
        const float* bp = B + (long long)(colBlock + m) * K + (k0 + kq);
        float4 bv = *(const float4*)bp;
        *(unsigned int*)&Bs[m][kq]     = pack2bf(bv.x, bv.y);
        *(unsigned int*)&Bs[m][kq + 2] = pack2bf(bv.z, bv.w);
      }
    } else {        // B is K x M: contiguous along m, scatter-transpose into LDS
#pragma unroll
      for (int it = 0; it < 2; ++it) {
        int j  = it * 256 + tid;      // 0..511 quads
        int k  = j >> 4;              // 0..31
        int mq = (j & 15) * 4;        // 0,4,...,60
        const float* bp = B + (long long)(k0 + k) * M + (colBlock + mq);
        float4 bv = *(const float4*)bp;
        Bs[mq + 0][k] = f2bf(bv.x);
        Bs[mq + 1][k] = f2bf(bv.y);
        Bs[mq + 2][k] = f2bf(bv.z);
        Bs[mq + 3][k] = f2bf(bv.w);
      }
    }
    __syncthreads();

    // ---- build fragments: A and B both use contiguous k = 16*g + s ----
    FragBF a[2], b[2];
#pragma unroll
    for (int i = 0; i < 2; ++i) {
      const unsigned short* arow = &As[wRow + i * 16 + l15][0];
#pragma unroll
      for (int v = 0; v < 4; ++v)
        a[i].u[v] = *(const unsigned int*)&arow[g * 16 + 2 * v];
#pragma unroll
      for (int v = 0; v < 4; ++v)
        a[i].u[4 + v] = *(const unsigned int*)&arow[g * 16 + 8 + 2 * v];
      const unsigned short* brow = &Bs[wCol + i * 16 + l15][0];
#pragma unroll
      for (int v = 0; v < 8; ++v)
        b[i].u[v] = *(const unsigned int*)&brow[g * 16 + 2 * v];
    }

#pragma unroll
    for (int i = 0; i < 2; ++i)
#pragma unroll
      for (int j = 0; j < 2; ++j)
        acc[i][j].v = __builtin_amdgcn_wmma_f32_16x16x32_bf16(
            false, a[i].v, false, b[j].v, (short)0, acc[i][j].v, false, false);

    __syncthreads();
  }

  // ---- epilogue: C layout: VGPR r -> row = r + g*8 ; col = lane&15 ----
#pragma unroll
  for (int i = 0; i < 2; ++i) {
#pragma unroll
    for (int j = 0; j < 2; ++j) {
      const int col = colBlock + wCol + j * 16 + l15;
#pragma unroll
      for (int r = 0; r < 8; ++r) {
        int row = rowBlock + wRow + i * 16 + r + g * 8;
        if (row < N) {
          float v = acc[i][j].f[r];
          long long idx = (long long)row * M + col;
          if (Cin)   v += Cin[idx];
          if (bias1) v += bias1[col];
          if (bias2) v += bias2[col];
          C[idx] = v;
        }
      }
    }
  }
}

// ---------------- helper kernels ----------------
__global__ void zero_kernel(float* p, long long n) {
  long long i = (long long)blockIdx.x * blockDim.x + threadIdx.x;
  for (; i < n; i += (long long)gridDim.x * blockDim.x) p[i] = 0.0f;
}

__global__ void deg_kernel(const int* __restrict__ dst, float* deg, int E) {
  int e = blockIdx.x * blockDim.x + threadIdx.x;
  if (e < E) atomicAdd(&deg[dst[e]], 1.0f);
}

__global__ void dinv_kernel(float* d, int n) {  // in-place: deg -> rsqrt(deg+1)
  int i = blockIdx.x * blockDim.x + threadIdx.x;
  if (i < n) d[i] = rsqrtf(d[i] + 1.0f);
}

__global__ void norme_kernel(const int* __restrict__ src, const int* __restrict__ dst,
                             const float* __restrict__ dinv, float* ne, int E) {
  int e = blockIdx.x * blockDim.x + threadIdx.x;
  if (e < E) ne[e] = dinv[src[e]] * dinv[dst[e]];
}

// one wave per edge: 64 channels, 2 per lane, atomic scatter into agg (L2-resident)
__global__ void edge_agg_kernel(const float* __restrict__ xw,
                                const int* __restrict__ src, const int* __restrict__ dst,
                                const float* __restrict__ ne, float* agg, int E) {
  int gw = (int)(((long long)blockIdx.x * blockDim.x + threadIdx.x) >> 5);
  int lane = threadIdx.x & 31;
  if (gw >= E) return;
  int s = src[gw], d = dst[gw];
  float w = ne[gw];
  const float2 v = ((const float2*)(xw + (long long)s * 64))[lane];
  float* out = agg + (long long)d * 64 + 2 * lane;
  atomicAdd(out,     v.x * w);
  atomicAdd(out + 1, v.y * w);
}

// x = agg + xw*dinv^2 + bias (in-place into agg), accumulate per-channel sum/sumsq
__global__ __launch_bounds__(256) void combine_stats_kernel(
    const float* __restrict__ xw, float* __restrict__ agg,
    const float* __restrict__ dinv, const float* __restrict__ bias,
    float* gsum, float* gsq, int N) {
  __shared__ float ssum[4][64], ssq[4][64];
  int tid = threadIdx.x;
  int c = tid & 63, grp = tid >> 6;
  int base = blockIdx.x * 64;
  float s = 0.0f, q = 0.0f;
  float bc = bias[c];
  for (int i = 0; i < 16; ++i) {
    int n = base + grp + 4 * i;
    if (n < N) {
      float di = dinv[n];
      long long idx = (long long)n * 64 + c;
      float v = agg[idx] + xw[idx] * (di * di) + bc;
      agg[idx] = v;
      s += v; q += v * v;
    }
  }
  ssum[grp][c] = s; ssq[grp][c] = q;
  __syncthreads();
  if (grp == 0) {
    atomicAdd(&gsum[c], ssum[0][c] + ssum[1][c] + ssum[2][c] + ssum[3][c]);
    atomicAdd(&gsq[c],  ssq[0][c]  + ssq[1][c]  + ssq[2][c]  + ssq[3][c]);
  }
}

__global__ void bn_apply_kernel(const float* __restrict__ xpre,
                                const float* __restrict__ gsum, const float* __restrict__ gsq,
                                const float* __restrict__ gam, const float* __restrict__ bet,
                                float* h, int N) {
  long long i = (long long)blockIdx.x * blockDim.x + threadIdx.x;
  if (i >= (long long)N * 64) return;
  int c = (int)(i & 63);
  float mean = gsum[c] / (float)N;
  float var  = gsq[c] / (float)N - mean * mean;
  h[i] = (xpre[i] - mean) * rsqrtf(var + BN_EPS) * gam[c] + bet[c];
}

__global__ void lstm_cell_kernel(const float* __restrict__ gates,
                                 float* __restrict__ h, float* __restrict__ c, int N) {
  long long i = (long long)blockIdx.x * blockDim.x + threadIdx.x;
  if (i >= (long long)N * LSTMH) return;
  int n = (int)(i / LSTMH), k = (int)(i % LSTMH);
  const float* gp = gates + (long long)n * NGATE;
  float ig = sigm(gp[k]);
  float fg = sigm(gp[LSTMH + k]);
  float gg = tanhf(gp[2 * LSTMH + k]);
  float og = sigm(gp[3 * LSTMH + k]);
  float cv = fg * c[i] + ig * gg;
  c[i] = cv;
  h[i] = og * tanhf(cv);
}

__global__ void att_contrib_kernel(const float* __restrict__ h, const float* __restrict__ attw,
                                   int woff, float* alpha, int t, int N) {
  int n = blockIdx.x * blockDim.x + threadIdx.x;
  if (n >= N) return;
  const float* hp = h + (long long)n * LSTMH;
  float s = 0.0f;
#pragma unroll 8
  for (int k = 0; k < LSTMH; ++k) s += hp[k] * attw[woff + k];
  alpha[n * 3 + t] += s;
}

__global__ void softmax3_kernel(float* alpha, int N) {
  int n = blockIdx.x * blockDim.x + threadIdx.x;
  if (n >= N) return;
  float a0 = alpha[n * 3], a1 = alpha[n * 3 + 1], a2 = alpha[n * 3 + 2];
  float m = fmaxf(a0, fmaxf(a1, a2));
  float e0 = __expf(a0 - m), e1 = __expf(a1 - m), e2 = __expf(a2 - m);
  float inv = 1.0f / (e0 + e1 + e2);
  alpha[n * 3] = e0 * inv; alpha[n * 3 + 1] = e1 * inv; alpha[n * 3 + 2] = e2 * inv;
}

__global__ void counts_kernel(const int* __restrict__ batch, float* counts, int N) {
  int n = blockIdx.x * blockDim.x + threadIdx.x;
  if (n < N) atomicAdd(&counts[batch[n]], 1.0f);
}

__global__ void outpool_kernel(const float* __restrict__ alpha,
                               const float* __restrict__ h0, const float* __restrict__ h1,
                               const float* __restrict__ h2, const int* __restrict__ batch,
                               float* pooled, int N) {
  long long i = (long long)blockIdx.x * blockDim.x + threadIdx.x;
  if (i >= (long long)N * 64) return;
  int n = (int)(i >> 6), c = (int)(i & 63);
  float v = alpha[n * 3] * h0[i] + alpha[n * 3 + 1] * h1[i] + alpha[n * 3 + 2] * h2[i];
  atomicAdd(&pooled[batch[n] * 64 + c], v);
}

__global__ void finalize_kernel(const float* __restrict__ pooled,
                                const float* __restrict__ counts, float* out) {
  int i = blockIdx.x * blockDim.x + threadIdx.x;
  if (i < NG * 64) out[i] = pooled[i] / fmaxf(counts[i >> 6], 1.0f);
}

// ---------------- host orchestration ----------------
static inline dim3 gemm_grid(int N, int M) { return dim3((N + TILE_N - 1) / TILE_N, M / TILE_M); }

extern "C" void kernel_launch(void* const* d_in, const int* in_sizes, int n_in,
                              void* d_out, int out_size, void* d_ws, size_t ws_size,
                              hipStream_t stream) {
  const float* x     = (const float*)d_in[0];
  const int*   eidx  = (const int*)d_in[1];
  const int*   src   = eidx;
  const int*   dst   = eidx + NE;
  const int*   batch = (const int*)d_in[2];
  const float* W[3]  = {(const float*)d_in[3], (const float*)d_in[4], (const float*)d_in[5]};
  const float* bL[3] = {(const float*)d_in[6], (const float*)d_in[7], (const float*)d_in[8]};
  const float* bng[3]= {(const float*)d_in[9], (const float*)d_in[10], (const float*)d_in[11]};
  const float* bnb[3]= {(const float*)d_in[12],(const float*)d_in[13],(const float*)d_in[14]};
  const float* Wih_f = (const float*)d_in[15];   // (384,64) row-major => transB GEMM
  const float* Whh_f = (const float*)d_in[16];   // (384,96)
  const float* bih_f = (const float*)d_in[17];
  const float* bhh_f = (const float*)d_in[18];
  const float* Wih_b = (const float*)d_in[19];
  const float* Whh_b = (const float*)d_in[20];
  const float* bih_b = (const float*)d_in[21];
  const float* bhh_b = (const float*)d_in[22];
  const float* attw  = (const float*)d_in[23];   // (192,)
  float* out = (float*)d_out;

  // workspace layout (floats)
  float* ws = (float*)d_ws;
  size_t o = 0;
  float* dinv   = ws + o; o += NN;
  float* ne     = ws + o; o += NE;
  float* xw     = ws + o; o += (size_t)NN * 64;
  float* agg    = ws + o; o += (size_t)NN * 64;     // becomes pre-BN x in-place
  float* h[3];
  for (int i = 0; i < 3; ++i) { h[i] = ws + o; o += (size_t)NN * 64; }
  float* gates  = ws + o; o += (size_t)NN * NGATE;
  float* hst    = ws + o; o += (size_t)NN * LSTMH;
  float* cst    = ws + o; o += (size_t)NN * LSTMH;
  float* alpha  = ws + o; o += (size_t)NN * 3;
  float* gsum   = ws + o; o += 64;
  float* gsq    = ws + o; o += 64;
  float* pooled = ws + o; o += NG * 64;
  float* counts = ws + o; o += NG;

  const int T = 256;
  auto blocks = [](long long n, int t) { return (int)((n + t - 1) / t); };

  // ---- degrees / norms ----
  zero_kernel<<<1024, T, 0, stream>>>(dinv, NN);
  deg_kernel<<<blocks(NE, T), T, 0, stream>>>(dst, dinv, NE);
  dinv_kernel<<<blocks(NN, T), T, 0, stream>>>(dinv, NN);
  norme_kernel<<<blocks(NE, T), T, 0, stream>>>(src, dst, dinv, ne, NE);

  // ---- 3 GCN layers ----
  const float* cur = x;
  int K = INC;
  for (int i = 0; i < 3; ++i) {
    gemm_bf16_wmma<<<gemm_grid(NN, 64), 256, 0, stream>>>(
        cur, W[i], nullptr, nullptr, nullptr, xw, NN, K, 64, 0);
    zero_kernel<<<4096, T, 0, stream>>>(agg, (long long)NN * 64);
    edge_agg_kernel<<<blocks((long long)NE * 32, T), T, 0, stream>>>(xw, src, dst, ne, agg, NE);
    zero_kernel<<<1, 128, 0, stream>>>(gsum, 64);
    zero_kernel<<<1, 128, 0, stream>>>(gsq, 64);
    combine_stats_kernel<<<blocks(NN, 64), 256, 0, stream>>>(xw, agg, dinv, bL[i], gsum, gsq, NN);
    bn_apply_kernel<<<blocks((long long)NN * 64, T), T, 0, stream>>>(
        agg, gsum, gsq, bng[i], bnb[i], h[i], NN);
    cur = h[i];
    K = HIDC;
  }

  // ---- bidirectional LSTM over the 3 layer outputs ----
  zero_kernel<<<2048, T, 0, stream>>>(alpha, (long long)NN * 3);

  // forward
  zero_kernel<<<4096, T, 0, stream>>>(hst, (long long)NN * LSTMH);
  zero_kernel<<<4096, T, 0, stream>>>(cst, (long long)NN * LSTMH);
  for (int t = 0; t < 3; ++t) {
    gemm_bf16_wmma<<<gemm_grid(NN, NGATE), 256, 0, stream>>>(
        h[t], Wih_f, bih_f, bhh_f, nullptr, gates, NN, OUTC, NGATE, 1);
    gemm_bf16_wmma<<<gemm_grid(NN, NGATE), 256, 0, stream>>>(
        hst, Whh_f, nullptr, nullptr, gates, gates, NN, LSTMH, NGATE, 1);
    lstm_cell_kernel<<<blocks((long long)NN * LSTMH, T), T, 0, stream>>>(gates, hst, cst, NN);
    att_contrib_kernel<<<blocks(NN, T), T, 0, stream>>>(hst, attw, 0, alpha, t, NN);
  }
  // backward
  zero_kernel<<<4096, T, 0, stream>>>(hst, (long long)NN * LSTMH);
  zero_kernel<<<4096, T, 0, stream>>>(cst, (long long)NN * LSTMH);
  for (int t = 2; t >= 0; --t) {
    gemm_bf16_wmma<<<gemm_grid(NN, NGATE), 256, 0, stream>>>(
        h[t], Wih_b, bih_b, bhh_b, nullptr, gates, NN, OUTC, NGATE, 1);
    gemm_bf16_wmma<<<gemm_grid(NN, NGATE), 256, 0, stream>>>(
        hst, Whh_b, nullptr, nullptr, gates, gates, NN, LSTMH, NGATE, 1);
    lstm_cell_kernel<<<blocks((long long)NN * LSTMH, T), T, 0, stream>>>(gates, hst, cst, NN);
    att_contrib_kernel<<<blocks(NN, T), T, 0, stream>>>(hst, attw, LSTMH, alpha, t, NN);
  }

  // ---- attention softmax + weighted sum + graph pooling ----
  softmax3_kernel<<<blocks(NN, T), T, 0, stream>>>(alpha, NN);
  zero_kernel<<<1, 256, 0, stream>>>(pooled, NG * 64);
  zero_kernel<<<1, 64, 0, stream>>>(counts, NG);
  counts_kernel<<<blocks(NN, T), T, 0, stream>>>(batch, counts, NN);
  outpool_kernel<<<blocks((long long)NN * 64, T), T, 0, stream>>>(
      alpha, h[0], h[1], h[2], batch, pooled, NN);
  finalize_kernel<<<blocks(NG * 64, T), T, 0, stream>>>(pooled, counts, out);
}
